// NeuronFlux2SingleStreamBlock_16020228014992
// MI455X (gfx1250) — matrix operand order repaired and checked by reading, not compiled, not run on hardware
//
#include <hip/hip_runtime.h>
#include <hip/hip_bf16.h>
#include <cstdint>
#include <cstddef>

// ---------------- problem constants ----------------
#define S_LEN   4096
#define DIM_    3072
#define HEADS_  24
#define HD_     128
#define MLP_    9216
#define NFUSED_ 27648          // 3*DIM + 2*MLP
#define EPS_    1e-6f

typedef __attribute__((ext_vector_type(8)))  float          v8f;
typedef __attribute__((ext_vector_type(16))) __bf16         v16bf;
typedef __attribute__((ext_vector_type(8)))  unsigned short us8;
typedef __attribute__((ext_vector_type(16))) unsigned short us16;
typedef unsigned short bfraw;

// ---------------- helpers ----------------
static __device__ __forceinline__ bfraw f2bf(float f) {
    unsigned u = __builtin_bit_cast(unsigned, f);
    unsigned r = u + 0x7FFFu + ((u >> 16) & 1u);   // RNE
    return (bfraw)(r >> 16);
}
static __device__ __forceinline__ float bf2f(bfraw h) {
    unsigned u = ((unsigned)h) << 16;
    return __builtin_bit_cast(float, u);
}
// Build a 32B WMMA fragment from two 16B LDS loads (all addresses 16B aligned).
static __device__ __forceinline__ v16bf load_frag(const bfraw* p0, const bfraw* p1) {
    us8 lo = *(const us8*)p0;
    us8 hi = *(const us8*)p1;
    us16 r = __builtin_shufflevector(lo, hi, 0,1,2,3,4,5,6,7,8,9,10,11,12,13,14,15);
    return __builtin_bit_cast(v16bf, r);
}
static __device__ __forceinline__ v8f wmma_bf16(v16bf a, v16bf b, v8f c) {
    return __builtin_amdgcn_wmma_f32_16x16x32_bf16(false, a, false, b, (short)0, c, false, false);
}
// CDNA5 async global->LDS copy: 16B per lane, tracked by ASYNCcnt (no VGPR data path).
static __device__ __forceinline__ void async_copy16(bfraw* lds_dst, const bfraw* gsrc) {
    unsigned lds = (unsigned)(uintptr_t)lds_dst;                 // addr[31:0] = LDS offset
    unsigned long long ga = (unsigned long long)(uintptr_t)gsrc;
    asm volatile("global_load_async_to_lds_b128 %0, %1, off"
                 :: "v"(lds), "v"(ga) : "memory");
}
static __device__ __forceinline__ void wait_async() {
    asm volatile("s_wait_asynccnt 0x0" ::: "memory");
}

// ---------------- kernel 1: LayerNorm + modulation -> bf16 ----------------
__global__ __launch_bounds__(256)
void ln_mod_kernel(const float* __restrict__ X, const float* __restrict__ temb,
                   bfraw* __restrict__ H)
{
    const int s = blockIdx.x, tid = threadIdx.x;
    __shared__ float red[16];
    float sum = 0.f, ss = 0.f;
    for (int i = tid; i < DIM_; i += 256) {
        float x = X[(size_t)s * DIM_ + i];
        sum += x; ss += x * x;
    }
    #pragma unroll
    for (int o = 16; o > 0; o >>= 1) {
        sum += __shfl_xor(sum, o, 32);
        ss  += __shfl_xor(ss,  o, 32);
    }
    if ((tid & 31) == 0) { red[tid >> 5] = sum; red[8 + (tid >> 5)] = ss; }
    __syncthreads();
    if (tid < 32) {
        float a = (tid < 8) ? red[tid]     : 0.f;
        float b = (tid < 8) ? red[8 + tid] : 0.f;
        #pragma unroll
        for (int o = 4; o > 0; o >>= 1) {
            a += __shfl_xor(a, o, 32);
            b += __shfl_xor(b, o, 32);
        }
        if (tid == 0) { red[0] = a; red[1] = b; }
    }
    __syncthreads();
    const float mu = red[0] * (1.0f / DIM_);
    const float var = red[1] * (1.0f / DIM_) - mu * mu;
    const float rs = rsqrtf(var + EPS_);
    for (int i = tid; i < DIM_; i += 256) {
        float x = X[(size_t)s * DIM_ + i];
        float h = (x - mu) * rs * (1.f + temb[DIM_ + i]) + temb[i];
        H[(size_t)s * DIM_ + i] = f2bf(h);
    }
}

// ---------------- kernel 2: weight convert f32[K,N] -> bf16[N,K] (once per launch) ----------------
__global__ __launch_bounds__(256)
void wt_convert_kernel(const float* __restrict__ W, bfraw* __restrict__ Wt, int K, int N)
{
    __shared__ bfraw sT[64][72];       // [n][k], 144B row stride (16B aligned)
    const int k0 = blockIdx.x * 64, n1 = blockIdx.y * 64;
    const int tid = threadIdx.x;
    {   // read 64x64 f32 tile (coalesced along N), convert + transpose into LDS
        const int kk = tid >> 2, nn0 = (tid & 3) * 16;
        const float* g = W + (size_t)(k0 + kk) * N + n1 + nn0;
        #pragma unroll
        for (int j = 0; j < 16; ++j) sT[nn0 + j][kk] = f2bf(g[j]);
    }
    __syncthreads();
    {   // vector-write bf16 rows of Wt (contiguous along K)
        const int nn = tid >> 2, kk0 = (tid & 3) * 16;
        bfraw* o = Wt + (size_t)(n1 + nn) * K + k0 + kk0;
        *(us8*)o       = *(const us8*)&sT[nn][kk0];
        *(us8*)(o + 8) = *(const us8*)&sT[nn][kk0 + 8];
    }
}

// ---------------- kernel 3: WMMA GEMM  C[M,N] = A[M,K](bf16) * Bt[N,K](bf16) ----------------
// BM=256, BN=128, BK=32, 256 threads = 8 waves (4x2), each wave a 64x64 tile
// (4x4 16x16 accum tiles -> 16 WMMA per K-step). Double-buffered LDS fed by
// GLOBAL_LOAD_ASYNC_TO_LDS_B128.
// EPI 0: store bf16   EPI 1: store f32   EPI 2: out = resid + gate*(tmp + acc) (f32)
template <int EPI>
__global__ __launch_bounds__(256)
void wmma_gemm(const bfraw* __restrict__ A, const bfraw* __restrict__ Bt,
               void* __restrict__ Cout, int M, int N, int K,
               const float* __restrict__ tmp, const float* __restrict__ resid,
               const float* __restrict__ gate)
{
    __shared__ bfraw sA[2][256][40];   // [m][k]
    __shared__ bfraw sB[2][128][40];   // [n][k]
    const int tid  = threadIdx.x;
    const int lane = tid & 31, wave = tid >> 5;
    const int wm = wave >> 1, wn = wave & 1;
    const int m0 = blockIdx.y * 256, n0 = blockIdx.x * 128;
    const int l15 = lane & 15, lh = lane >> 4;
    const int k0a = lh * 8;    // ISA 16-bit A layout half-wave k offset
    const int k0b = lh * 8;    // B layout: same half-wave k offset as A (K 8h..8h+7, 16+8h..16+8h+7)

    v8f acc[4][4];
    #pragma unroll
    for (int i = 0; i < 4; ++i)
        #pragma unroll
        for (int j = 0; j < 4; ++j)
            #pragma unroll
            for (int e = 0; e < 8; ++e) acc[i][j][e] = 0.f;

    // A: 256x32 tile, 32 elts/thread (4 async b128); B: 128x32 tile, 16 elts/thread (2 async b128)
    const int arow = tid;
    const int brow = tid >> 1, bc0 = (tid & 1) * 16;
    const bfraw* gA = A  + (size_t)(m0 + arow) * K;
    const bfraw* gB = Bt + (size_t)(n0 + brow) * K + bc0;

    auto stream_tile = [&](int k, int buf) {
        const bfraw* ga = gA + k;
        async_copy16(&sA[buf][arow][0],  ga);
        async_copy16(&sA[buf][arow][8],  ga + 8);
        async_copy16(&sA[buf][arow][16], ga + 16);
        async_copy16(&sA[buf][arow][24], ga + 24);
        const bfraw* gb = gB + k;
        async_copy16(&sB[buf][brow][bc0],     gb);
        async_copy16(&sB[buf][brow][bc0 + 8], gb + 8);
    };

    stream_tile(0, 0);
    wait_async();
    __syncthreads();

    int cur = 0;
    for (int k = 0; k < K; k += 32) {
        const int nxt = cur ^ 1;
        if (k + 32 < K) stream_tile(k + 32, nxt);   // overlap with compute

        v16bf af[4], bfr[4];
        #pragma unroll
        for (int i = 0; i < 4; ++i) {
            const bfraw* p = &sA[cur][wm * 64 + i * 16 + l15][k0a];
            af[i] = load_frag(p, p + 16);
        }
        #pragma unroll
        for (int j = 0; j < 4; ++j) {
            const bfraw* p = &sB[cur][wn * 64 + j * 16 + l15][k0b];
            bfr[j] = load_frag(p, p + 16);
        }
        #pragma unroll
        for (int i = 0; i < 4; ++i)
            #pragma unroll
            for (int j = 0; j < 4; ++j)
                acc[i][j] = wmma_bf16(af[i], bfr[j], acc[i][j]);
        wait_async();
        __syncthreads();
        cur = nxt;
    }

    #pragma unroll
    for (int i = 0; i < 4; ++i)
        #pragma unroll
        for (int j = 0; j < 4; ++j)
            #pragma unroll
            for (int r = 0; r < 8; ++r) {
                const int row = m0 + wm * 64 + i * 16 + lh * 8 + r;
                const int col = n0 + wn * 64 + j * 16 + l15;
                const size_t idx = (size_t)row * N + col;
                const float v = acc[i][j][r];
                if (EPI == 0)      ((bfraw*)Cout)[idx] = f2bf(v);
                else if (EPI == 1) ((float*)Cout)[idx] = v;
                else               ((float*)Cout)[idx] =
                        resid[idx] + gate[col] * (tmp[idx] + v);
            }
}

// ---------------- kernel 4: RMSNorm + RoPE, scatter to head-major Q/K and d-major V ----------------
__global__ __launch_bounds__(128)
void qkv_post_kernel(const bfraw* __restrict__ QKV,
                     const float* __restrict__ rcos, const float* __restrict__ rsin,
                     const float* __restrict__ nqw,  const float* __restrict__ nkw,
                     bfraw* __restrict__ qh, bfraw* __restrict__ kh, bfraw* __restrict__ vt)
{
    const int s = blockIdx.x, head = blockIdx.y, d = threadIdx.x;
    __shared__ float red[8];
    const size_t base = (size_t)s * NFUSED_ + (size_t)head * HD_;
    float q = bf2f(QKV[base + d]);
    float k = bf2f(QKV[base + DIM_ + d]);
    bfraw v = QKV[base + 2 * DIM_ + d];

    float sq = q * q, sk = k * k;
    #pragma unroll
    for (int o = 16; o > 0; o >>= 1) {
        sq += __shfl_xor(sq, o, 32);
        sk += __shfl_xor(sk, o, 32);
    }
    if ((d & 31) == 0) { red[d >> 5] = sq; red[4 + (d >> 5)] = sk; }
    __syncthreads();
    const float tq = red[0] + red[1] + red[2] + red[3];
    const float tk = red[4] + red[5] + red[6] + red[7];
    q *= rsqrtf(tq * (1.0f / HD_) + EPS_) * nqw[d];
    k *= rsqrtf(tk * (1.0f / HD_) + EPS_) * nkw[d];

    const float qp = __shfl_xor(q, 1, 32);
    const float kp = __shfl_xor(k, 1, 32);
    const float qrot = (d & 1) ? qp : -qp;
    const float krot = (d & 1) ? kp : -kp;
    const float c  = rcos[(size_t)s * HD_ + d];
    const float sn = rsin[(size_t)s * HD_ + d];
    const size_t o = ((size_t)head * S_LEN + s) * HD_ + d;
    qh[o] = f2bf(q * c + qrot * sn);
    kh[o] = f2bf(k * c + krot * sn);
    vt[((size_t)head * HD_ + d) * S_LEN + s] = v;   // V stored transposed: [head][d][S]
}

// ---------------- kernel 5: SwiGLU ----------------
__global__ __launch_bounds__(256)
void swiglu_kernel(const bfraw* __restrict__ QKV, bfraw* __restrict__ out, int n)
{
    const int idx = blockIdx.x * 256 + threadIdx.x;
    if (idx >= n) return;
    const int s = idx / MLP_, j = idx - s * MLP_;
    const size_t base = (size_t)s * NFUSED_ + 3 * DIM_;
    const float a = bf2f(QKV[base + j]);
    const float b = bf2f(QKV[base + MLP_ + j]);
    const float si = a / (1.f + __expf(-a));
    out[idx] = f2bf(si * b);
}

// ---------------- kernel 6: flash attention (async double-buffered K/Vt, WMMA QK^T & PV) ----------------
// grid = (S/128, HEADS), 256 threads = 8 waves; each wave owns 16 query rows.
__global__ __launch_bounds__(256)
void flash_attn_kernel(const bfraw* __restrict__ Qh, const bfraw* __restrict__ Kh,
                       const bfraw* __restrict__ Vt, bfraw* __restrict__ Out)
{
    __shared__ bfraw sQ[128][136];        // [q][d]
    __shared__ bfraw sK[2][32][136];      // [kcol][d]  (B of QK^T: contiguous along d)
    __shared__ bfraw sVt[2][128][40];     // [d][krow]  (B of PV: contiguous along k)
    __shared__ bfraw sP[8][16][40];       // per-wave P staging (C-layout -> A-layout)

    const int head = blockIdx.y;
    const int q0 = blockIdx.x * 128;
    const int tid = threadIdx.x;
    const int lane = tid & 31, wave = tid >> 5;
    const int l15 = lane & 15, lh = lane >> 4;
    const int k0a = lh * 8, k0b = lh * 8;
    const float SCALE = 0.08838834764831845f;   // 1/sqrt(128)

    const bfraw* Qp  = Qh + (size_t)head * S_LEN * HD_;
    const bfraw* Kp  = Kh + (size_t)head * S_LEN * HD_;
    const bfraw* Vtp = Vt + (size_t)head * HD_ * S_LEN;

    // K/Vt block streamers (16 elts / thread each)
    const int krow = tid >> 3, kc0 = (tid & 7) * 16;
    const int vrow = tid >> 1, vc0 = (tid & 1) * 16;
    auto load_kv = [&](int kb, int buf) {
        const bfraw* gk = Kp + (size_t)(kb * 32 + krow) * HD_ + kc0;
        async_copy16(&sK[buf][krow][kc0],     gk);
        async_copy16(&sK[buf][krow][kc0 + 8], gk + 8);
        const bfraw* gv = Vtp + (size_t)vrow * S_LEN + kb * 32 + vc0;
        async_copy16(&sVt[buf][vrow][vc0],     gv);
        async_copy16(&sVt[buf][vrow][vc0 + 8], gv + 8);
    };

    {   // async-load Q tile 128x128 (64 elts / thread) + first K/V block
        const int row = tid >> 1, c0q = (tid & 1) * 64;
        const bfraw* g = Qp + (size_t)(q0 + row) * HD_ + c0q;
        #pragma unroll
        for (int j = 0; j < 64; j += 8)
            async_copy16(&sQ[row][c0q + j], g + j);
        load_kv(0, 0);
        wait_async();
        __syncthreads();
    }

    float mstate[8], lstate[8];
    v8f oacc[8];
    #pragma unroll
    for (int r = 0; r < 8; ++r) { mstate[r] = -3.0e38f; lstate[r] = 0.f; }
    #pragma unroll
    for (int nd = 0; nd < 8; ++nd)
        #pragma unroll
        for (int e = 0; e < 8; ++e) oacc[nd][e] = 0.f;

    const int NKB = S_LEN / 32;
    int cur = 0;
    for (int kb = 0; kb < NKB; ++kb) {
        if (kb + 1 < NKB) load_kv(kb + 1, cur ^ 1);

        // ---- scores: S(16x32) = Q(16x128) K^T, two 16x16 N-tiles ----
        v8f sc0, sc1;
        #pragma unroll
        for (int e = 0; e < 8; ++e) { sc0[e] = 0.f; sc1[e] = 0.f; }
        #pragma unroll
        for (int dc = 0; dc < 4; ++dc) {
            const bfraw* pq = &sQ[wave * 16 + l15][dc * 32 + k0a];
            const v16bf aq = load_frag(pq, pq + 16);
            const bfraw* pk0 = &sK[cur][l15][dc * 32 + k0b];
            const bfraw* pk1 = &sK[cur][16 + l15][dc * 32 + k0b];
            sc0 = wmma_bf16(aq, load_frag(pk0, pk0 + 16), sc0);
            sc1 = wmma_bf16(aq, load_frag(pk1, pk1 + 16), sc1);
        }

        // ---- online softmax (rows live in 16-lane halves; xor masks 8..1) ----
        #pragma unroll
        for (int r = 0; r < 8; ++r) {
            float s0 = sc0[r] * SCALE, s1 = sc1[r] * SCALE;
            float mv = fmaxf(s0, s1);
            #pragma unroll
            for (int o = 8; o > 0; o >>= 1) mv = fmaxf(mv, __shfl_xor(mv, o, 32));
            const float mn = fmaxf(mstate[r], mv);
            const float alpha = __expf(mstate[r] - mn);
            mstate[r] = mn;
            const float p0 = __expf(s0 - mn), p1 = __expf(s1 - mn);
            sc0[r] = p0; sc1[r] = p1;
            float ps = p0 + p1;
            #pragma unroll
            for (int o = 8; o > 0; o >>= 1) ps += __shfl_xor(ps, o, 32);
            lstate[r] = lstate[r] * alpha + ps;
            #pragma unroll
            for (int nd = 0; nd < 8; ++nd) oacc[nd][r] *= alpha;
        }

        // ---- P: C-layout -> A-layout bounce through per-wave LDS ----
        #pragma unroll
        for (int r = 0; r < 8; ++r) {
            sP[wave][lh * 8 + r][l15]      = f2bf(sc0[r]);
            sP[wave][lh * 8 + r][16 + l15] = f2bf(sc1[r]);
        }
        {
            const bfraw* pp = &sP[wave][l15][k0a];
            const v16bf ap = load_frag(pp, pp + 16);
            #pragma unroll
            for (int nd = 0; nd < 8; ++nd) {
                const bfraw* pv = &sVt[cur][nd * 16 + l15][k0b];
                oacc[nd] = wmma_bf16(ap, load_frag(pv, pv + 16), oacc[nd]);
            }
        }
        wait_async();
        __syncthreads();
        cur ^= 1;
    }

    float linv[8];
    #pragma unroll
    for (int r = 0; r < 8; ++r) linv[r] = 1.0f / lstate[r];
    #pragma unroll
    for (int nd = 0; nd < 8; ++nd)
        #pragma unroll
        for (int r = 0; r < 8; ++r) {
            const int row = q0 + wave * 16 + lh * 8 + r;
            const int col = head * HD_ + nd * 16 + l15;
            Out[(size_t)row * DIM_ + col] = f2bf(oacc[nd][r] * linv[r]);
        }
}

// ---------------- launcher ----------------
extern "C" void kernel_launch(void* const* d_in, const int* in_sizes, int n_in,
                              void* d_out, int out_size, void* d_ws, size_t ws_size,
                              hipStream_t stream)
{
    (void)in_sizes; (void)n_in; (void)out_size; (void)ws_size;

    const float* hs    = (const float*)d_in[0];   // [1,4096,3072]
    const float* temb  = (const float*)d_in[1];   // [1,9216]
    const float* rcos  = (const float*)d_in[2];   // [4096,128]
    const float* rsin  = (const float*)d_in[3];   // [4096,128]
    const float* w_qkv = (const float*)d_in[4];   // [3072,27648]
    const float* w_oa  = (const float*)d_in[5];   // [3072,3072]
    const float* w_om  = (const float*)d_in[6];   // [9216,3072]
    const float* nqw   = (const float*)d_in[7];   // [128]
    const float* nkw   = (const float*)d_in[8];   // [128]
    float* out = (float*)d_out;

    // workspace carve (~723 MB total)
    char* w = (char*)d_ws;
    size_t off = 0;
    auto carve = [&](size_t bytes) -> char* {
        char* p = w + off;
        off += (bytes + 255) & ~(size_t)255;
        return p;
    };
    bfraw* h_bf    = (bfraw*)carve((size_t)S_LEN * DIM_ * 2);
    bfraw* qkvmlp  = (bfraw*)carve((size_t)S_LEN * NFUSED_ * 2);
    bfraw* qh      = (bfraw*)carve((size_t)HEADS_ * S_LEN * HD_ * 2);
    bfraw* kh      = (bfraw*)carve((size_t)HEADS_ * S_LEN * HD_ * 2);
    bfraw* vt      = (bfraw*)carve((size_t)HEADS_ * S_LEN * HD_ * 2);
    bfraw* attn    = (bfraw*)carve((size_t)S_LEN * DIM_ * 2);
    bfraw* mlpact  = (bfraw*)carve((size_t)S_LEN * MLP_ * 2);
    float* tmp     = (float*)carve((size_t)S_LEN * DIM_ * 4);
    bfraw* wqkv_t  = (bfraw*)carve((size_t)NFUSED_ * DIM_ * 2);   // [N][K]
    bfraw* woa_t   = (bfraw*)carve((size_t)DIM_ * DIM_ * 2);
    bfraw* wom_t   = (bfraw*)carve((size_t)DIM_ * MLP_ * 2);

    // 0. one-shot weight convert + transpose (f32[K,N] -> bf16[N,K])
    wt_convert_kernel<<<dim3(DIM_ / 64, NFUSED_ / 64), 256, 0, stream>>>(w_qkv, wqkv_t, DIM_, NFUSED_);
    wt_convert_kernel<<<dim3(DIM_ / 64, DIM_ / 64),    256, 0, stream>>>(w_oa,  woa_t,  DIM_, DIM_);
    wt_convert_kernel<<<dim3(MLP_ / 64, DIM_ / 64),    256, 0, stream>>>(w_om,  wom_t,  MLP_, DIM_);

    // 1. LayerNorm + modulation
    ln_mod_kernel<<<S_LEN, 256, 0, stream>>>(hs, temb, h_bf);

    // 2. fused QKV+MLP GEMM: [4096,3072] x [3072,27648] -> bf16
    wmma_gemm<0><<<dim3(NFUSED_ / 128, S_LEN / 256), 256, 0, stream>>>(
        h_bf, wqkv_t, qkvmlp, S_LEN, NFUSED_, DIM_, nullptr, nullptr, nullptr);

    // 3. RMSNorm + RoPE -> head-major q/k, d-major v
    qkv_post_kernel<<<dim3(S_LEN, HEADS_), 128, 0, stream>>>(
        qkvmlp, rcos, rsin, nqw, nkw, qh, kh, vt);

    // 4. SwiGLU
    {
        const int n = S_LEN * MLP_;
        swiglu_kernel<<<(n + 255) / 256, 256, 0, stream>>>(qkvmlp, mlpact, n);
    }

    // 5. flash attention
    flash_attn_kernel<<<dim3(S_LEN / 128, HEADS_), 256, 0, stream>>>(qh, kh, vt, attn);

    // 6. attn output projection -> tmp (f32)
    wmma_gemm<1><<<dim3(DIM_ / 128, S_LEN / 256), 256, 0, stream>>>(
        attn, woa_t, tmp, S_LEN, DIM_, DIM_, nullptr, nullptr, nullptr);

    // 7. mlp output projection + residual + gate -> out
    wmma_gemm<2><<<dim3(DIM_ / 128, S_LEN / 256), 256, 0, stream>>>(
        mlpact, wom_t, out, S_LEN, DIM_, MLP_, tmp, hs, temb + 2 * DIM_);
}
